// ConvLSTM_62096637165743
// MI455X (gfx1250) — matrix operand order repaired and checked by reading, not compiled, not run on hardware
//
#include <hip/hip_runtime.h>
#include <hip/hip_bf16.h>
#include <stdint.h>
#include <stddef.h>

// ---------------------------------------------------------------------------
// ConvLSTM on MI455X (gfx1250): implicit-GEMM conv via v_wmma_f32_16x16x32_bf16
// ---------------------------------------------------------------------------

typedef __attribute__((ext_vector_type(16))) __bf16 v16bf;
typedef __attribute__((ext_vector_type(8)))  __bf16 v8bf;
typedef __attribute__((ext_vector_type(8)))  float  v8f;

#define IN_CH   64
#define HID     96
#define BATCH   8
#define TSTEPS  16
#define HH      64
#define WW      64
#define KDIM    1440          // (IN_CH+HID)*9
#define KBLK    45            // KDIM/32
#define NPIX    32            // pixels per block tile (along W)
#define NWAVES  6             // one wave per 16-hidden-channel group
#define NTHREADS (NWAVES*32)  // 192

__device__ __forceinline__ float sigmoid_f(float x) {
    return 1.0f / (1.0f + __expf(-x));
}

__global__ void zero_u32(uint32_t* __restrict__ p, int n) {
    int i = blockIdx.x * blockDim.x + threadIdx.x;
    if (i < n) p[i] = 0u;
}

__global__ void convert_w(const float* __restrict__ W, __bf16* __restrict__ Wp, int n) {
    int i = blockIdx.x * blockDim.x + threadIdx.x;
    if (i < n) Wp[i] = (__bf16)W[i];
}

__global__ __launch_bounds__(NTHREADS)
void lstm_step(const float*  __restrict__ xseq,   // [B,T,64,64,64] f32
               const __bf16* __restrict__ Wp,     // [384,1440] bf16 (prepacked)
               const float*  __restrict__ bias,   // [384]
               const float*  __restrict__ wci,    // [96]
               const float*  __restrict__ wcf,    // [96]
               const float*  __restrict__ wco,    // [96]
               const __bf16* __restrict__ hprev,  // [B,96,64,64] bf16
               __bf16*       __restrict__ hnext,  // [B,96,64,64] bf16
               float*        __restrict__ cbuf,   // [B,96,64,64] f32 (in-place)
               float*        __restrict__ yout,   // [B,T,96,64,64] f32
               int t)
{
    // fragment-ready im2col patch: [KBLK][NPIX][32] bf16 = 92160 bytes
    extern __shared__ __bf16 lds[];

    const int tid  = threadIdx.x;
    const int x0   = blockIdx.x * NPIX;
    const int yrow = blockIdx.y;
    const int b    = blockIdx.z;

    // ---- stage im2col patch into LDS (convert f32 x -> bf16 on the fly) ----
    const int tile_elems = KDIM * NPIX;   // 46080
    for (int e = tid; e < tile_elems; e += NTHREADS) {
        const int n   = e & (NPIX - 1);   // pixel within tile (NPIX==32)
        const int k   = e >> 5;           // 0..1439 = ci*9 + kh*3 + kw
        const int ci  = k / 9;
        const int tap = k - ci * 9;
        const int kh  = tap / 3;
        const int kw  = tap - kh * 3;
        const int yy  = yrow + kh - 1;
        const int xx  = x0 + n + kw - 1;
        __bf16 v = (__bf16)0.0f;
        if (yy >= 0 && yy < HH && xx >= 0 && xx < WW) {
            if (ci < IN_CH) {
                float f = xseq[((((size_t)b * TSTEPS) + t) * IN_CH + ci) * (HH * WW)
                               + yy * WW + xx];
                v = (__bf16)f;
            } else {
                v = hprev[(((size_t)b * HID) + (ci - IN_CH)) * (HH * WW)
                          + yy * WW + xx];
            }
        }
        const int kb = k >> 5;
        const int ko = k & 31;
        lds[((size_t)kb * NPIX + n) * 32 + ko] = v;
    }
    __syncthreads();

    // ---- implicit GEMM: gates[384 x 32px] = W[384 x 1440] * patch[1440 x 32] ----
    const int wave = tid >> 5;        // hidden-channel group 0..5
    const int lane = tid & 31;
    const int m    = lane & 15;       // A row within 16 / B,C column
    const int half = lane >> 4;       // K-half select

    v8f acc[8] = {};                  // [gate i,f,g,o][pixel subtile 0,1]

    for (int kb = 0; kb < KBLK; ++kb) {
        // B fragments (32x16 bf16): lane holds col=(lane&15),
        // K = half*16 .. half*16+15 contiguous -> single 32B LDS read per frag.
        const __bf16* bbase = &lds[((size_t)kb * NPIX) * 32];
        v16bf bfrag0 = *(const v16bf*)(bbase + ((size_t)m * 32        + half * 16));
        v16bf bfrag1 = *(const v16bf*)(bbase + ((size_t)(m + 16) * 32 + half * 16));
#pragma unroll
        for (int g = 0; g < 4; ++g) {
            const int row = g * HID + wave * 16 + m;   // output channel
            const __bf16* ap = Wp + (size_t)row * KDIM + (size_t)kb * 32 + half * 16;
            v8bf alo = *(const v8bf*)ap;          // K = half*16 + 0..7
            v8bf ahi = *(const v8bf*)(ap + 8);    // K = half*16 + 8..15
            v16bf afrag = __builtin_shufflevector(alo, ahi,
                0, 1, 2, 3, 4, 5, 6, 7, 8, 9, 10, 11, 12, 13, 14, 15);
            acc[g * 2 + 0] = __builtin_amdgcn_wmma_f32_16x16x32_bf16(
                false, afrag, false, bfrag0, (short)0, acc[g * 2 + 0], false, false);
            acc[g * 2 + 1] = __builtin_amdgcn_wmma_f32_16x16x32_bf16(
                false, afrag, false, bfrag1, (short)0, acc[g * 2 + 1], false, false);
        }
    }

    // ---- peephole-LSTM pointwise update, entirely in registers ----
    // C/D layout: VGPR r, lane<16 -> M=r, lane>=16 -> M=r+8; N = lane&15.
#pragma unroll
    for (int sub = 0; sub < 2; ++sub) {
        const int px = x0 + sub * 16 + m;
#pragma unroll
        for (int r = 0; r < 8; ++r) {
            const int hc = wave * 16 + half * 8 + r;
            const size_t sidx = (((size_t)b * HID) + hc) * (HH * WW) + yrow * WW + px;
            const float cp = cbuf[sidx];
            const float gi = acc[0 * 2 + sub][r] + bias[hc];
            const float gf = acc[1 * 2 + sub][r] + bias[HID + hc];
            const float gg = acc[2 * 2 + sub][r] + bias[2 * HID + hc];
            const float go = acc[3 * 2 + sub][r] + bias[3 * HID + hc];
            const float iv = sigmoid_f(gi + wci[hc] * cp);
            const float fv = sigmoid_f(gf + wcf[hc] * cp);
            const float cn = fv * cp + iv * tanhf(gg);
            const float ov = sigmoid_f(go + wco[hc] * cn);
            const float hn = ov * tanhf(cn);
            cbuf[sidx]  = cn;            // in-place: this block owns these elems
            hnext[sidx] = (__bf16)hn;    // bf16 copy feeds next step's conv
            yout[((((size_t)b * TSTEPS) + t) * HID + hc) * (HH * WW)
                 + yrow * WW + px] = hn;
        }
    }
}

extern "C" void kernel_launch(void* const* d_in, const int* in_sizes, int n_in,
                              void* d_out, int out_size, void* d_ws, size_t ws_size,
                              hipStream_t stream) {
    (void)in_sizes; (void)n_in; (void)out_size; (void)ws_size;

    const float* x    = (const float*)d_in[0];  // x_seq
    const float* W    = (const float*)d_in[1];  // (384,160,3,3)
    const float* bias = (const float*)d_in[2];  // (384,)
    const float* wci  = (const float*)d_in[3];  // (96,1,1)
    const float* wcf  = (const float*)d_in[4];
    const float* wco  = (const float*)d_in[5];
    float* y = (float*)d_out;                   // (8,16,96,64,64) f32

    char* ws = (char*)d_ws;
    size_t off = 0;
    __bf16* Wp = (__bf16*)(ws + off); off += (size_t)4 * HID * KDIM * 2;            // 1.1 MB
    __bf16* h0 = (__bf16*)(ws + off); off += (size_t)BATCH * HID * HH * WW * 2;     // 6.3 MB
    __bf16* h1 = (__bf16*)(ws + off); off += (size_t)BATCH * HID * HH * WW * 2;     // 6.3 MB
    float*  c  = (float*)(ws + off);  off += (size_t)BATCH * HID * HH * WW * 4;     // 12.6 MB

    // zero initial state every call (deterministic; c is updated in place)
    {
        int nh = (BATCH * HID * HH * WW * 2) / 4;   // h0 as u32 words
        zero_u32<<<(nh + 255) / 256, 256, 0, stream>>>((uint32_t*)h0, nh);
        int nc = BATCH * HID * HH * WW;
        zero_u32<<<(nc + 255) / 256, 256, 0, stream>>>((uint32_t*)c, nc);
    }
    // prepack weights: f32 -> bf16 (flat layout already matches im2col k-order)
    {
        int n = 4 * HID * KDIM;
        convert_w<<<(n + 255) / 256, 256, 0, stream>>>(W, Wp, n);
    }

    const size_t ldsbytes = (size_t)KBLK * NPIX * 32 * sizeof(__bf16);  // 92160
    (void)hipFuncSetAttribute((const void*)lstm_step,
                              hipFuncAttributeMaxDynamicSharedMemorySize,
                              (int)ldsbytes);

    dim3 grid(WW / NPIX, HH, BATCH);   // (2, 64, 8)
    for (int t = 0; t < TSTEPS; ++t) {
        const __bf16* hp = (t & 1) ? h1 : h0;
        __bf16*       hn = (t & 1) ? h0 : h1;
        lstm_step<<<grid, NTHREADS, ldsbytes, stream>>>(
            x, Wp, bias, wci, wcf, wco, hp, hn, c, y, t);
    }
}
